// Soft_Attention_50826642981282
// MI455X (gfx1250) — matrix operand order repaired and checked by reading, not compiled, not run on hardware
//
#include <hip/hip_runtime.h>
#include <hip/hip_bf16.h>

// Soft attention (Bahdanau-style) fused kernel for MI455X (gfx1250).
// Dominant GEMM done with v_wmma_f32_16x16x32_bf16 (bf16 in, f32 acc).

typedef __bf16 bf16_t;
typedef bf16_t v16bf __attribute__((ext_vector_type(16)));
typedef float  v8f   __attribute__((ext_vector_type(8)));

#define B_    256
#define P_    196
#define DE_   2048
#define DD_   512
#define DA_   512
#define PT_   13      // ceil(196/16)
#define KSTEP 32      // WMMA K per instruction
#define KCHUNK 128    // K staged in LDS per barrier window (4 WMMA k-steps)
#define ASTRIDE 136   // LDS row stride in bf16 (128 + 8 pad, keeps 16B align)

static __device__ __forceinline__ unsigned short f2bf_bits(float f) {
  // round-to-nearest-even f32 -> bf16, pure integer path (always compiles)
  unsigned u = __builtin_bit_cast(unsigned, f);
  u += 0x7fffu + ((u >> 16) & 1u);
  return (unsigned short)(u >> 16);
}

static __device__ __forceinline__ unsigned int pack2bf(float x, float y) {
  return (unsigned int)f2bf_bits(x) | ((unsigned int)f2bf_bits(y) << 16);
}

union V16U { v16bf v; uint4 u[2]; };

// ---------------------------------------------------------------------------
// Kernel 0: W_enc [DE][DA] f32  ->  WT [DA][DE] bf16 (transpose + convert)
// ---------------------------------------------------------------------------
__global__ __launch_bounds__(256) void wenc_transpose_bf16(
    const float* __restrict__ W, unsigned short* __restrict__ WT) {
  __shared__ float tile[32][33];
  const int a0 = blockIdx.x * 32;
  const int k0 = blockIdx.y * 32;
  const int tx = threadIdx.x & 31;
  const int ty = threadIdx.x >> 5;   // 0..7
#pragma unroll
  for (int r = 0; r < 32; r += 8)
    tile[ty + r][tx] = W[(size_t)(k0 + ty + r) * DA_ + a0 + tx];
  __syncthreads();
#pragma unroll
  for (int r = 0; r < 32; r += 8)
    WT[(size_t)(a0 + ty + r) * DE_ + k0 + tx] = f2bf_bits(tile[tx][ty + r]);
}

// ---------------------------------------------------------------------------
// Kernel 1: pre[b][a] = b_dec[a] + b_enc[a] + dot(decoder_hidden[b], W_dec[:,a])
// ---------------------------------------------------------------------------
__global__ __launch_bounds__(256) void att2_pre_kernel(
    const float* __restrict__ dh, const float* __restrict__ Wd,
    const float* __restrict__ bd, const float* __restrict__ be,
    float* __restrict__ pre) {
  const int idx = blockIdx.x * 256 + threadIdx.x;   // < B_*DA_
  const int b = idx >> 9;
  const int a = idx & (DA_ - 1);
  const float* dhp = dh + (size_t)b * DD_;
  float s = bd[a] + be[a];
#pragma unroll 4
  for (int d = 0; d < DD_; ++d) s += dhp[d] * Wd[(size_t)d * DA_ + a];
  pre[idx] = s;
}

// ---------------------------------------------------------------------------
// Kernel 2: fused scores + softmax + context, one block per batch element.
// 256 threads = 8 waves (wave32). All waves share the p-tile (A in LDS),
// each wave owns 4 a-tiles of 16 columns (8*4*16 = 512 = DA).
// ---------------------------------------------------------------------------
__global__ __launch_bounds__(256) void attn_main(
    const float* __restrict__ enc, const unsigned short* __restrict__ WT,
    const float* __restrict__ pre, const float* __restrict__ Wfull,
    const float* __restrict__ bfull, float* __restrict__ ctx,
    float* __restrict__ alpha) {
  const int b    = blockIdx.x;
  const int tid  = (int)threadIdx.x;
  const int lane = tid & 31;
  const int w    = tid >> 5;      // wave id 0..7
  const int l15  = lane & 15;
  const int hi   = lane >> 4;     // 0 / 1

  __shared__ __align__(16) unsigned short sA[16 * ASTRIDE];  // 16 x 128 bf16
  __shared__ float sPart[8][16];
  __shared__ float sAtt[208];
  __shared__ float sRed[256];

  // Per-lane column constants: a = (4w+i)*16 + l15 (C layout: lanes = N cols)
  float av[4], wv[4];
#pragma unroll
  for (int i = 0; i < 4; ++i) {
    const int a = (w * 4 + i) * 16 + l15;
    av[i] = pre[(size_t)b * DA_ + a];
    wv[i] = Wfull[a];
  }
  const float bf_bias = bfull[0];
  const float* encB = enc + (size_t)b * P_ * DE_;

  // cooperative-loader coordinates: 8 f32 per thread (16*128 = 2048 = 256*8)
  const int ld_row = tid >> 4;          // 0..15
  const int ld_k   = (tid & 15) * 8;    // 0..120 step 8

  for (int pt = 0; pt < PT_; ++pt) {
    const int p0 = pt * 16;
    v8f acc[4] = {};

    for (int kb0 = 0; kb0 < DE_; kb0 += KCHUNK) {
      __syncthreads();  // protect LDS slab reuse
      {
        // cooperative load of 16 x 128 f32 -> bf16 in LDS (8 elems/thread)
        const int p = p0 + ld_row;
        float4 x = {0.f, 0.f, 0.f, 0.f}, y = {0.f, 0.f, 0.f, 0.f};
        if (p < P_) {
          const float4* src =
              (const float4*)(encB + (size_t)p * DE_ + kb0 + ld_k);
          x = src[0];
          y = src[1];
          if (kb0 + KCHUNK < DE_)
            __builtin_prefetch((const float*)src + KCHUNK, 0, 0);
        }
        uint4 pk;
        pk.x = pack2bf(x.x, x.y);
        pk.y = pack2bf(x.z, x.w);
        pk.z = pack2bf(y.x, y.y);
        pk.w = pack2bf(y.z, y.w);
        *(uint4*)&sA[ld_row * ASTRIDE + ld_k] = pk;
      }
      __syncthreads();

      // 4 WMMA k-steps per barrier window, no barriers inside
#pragma unroll
      for (int kk = 0; kk < 4; ++kk) {
        const int kb = kb0 + kk * KSTEP;

        // A fragment (16x32 bf16): lane<16 -> K {0..7,16..23}; lane>=16 -> +8
        V16U A;
        {
          const unsigned short* ap =
              &sA[l15 * ASTRIDE + kk * KSTEP + hi * 8];
          A.u[0] = *(const uint4*)ap;
          A.u[1] = *(const uint4*)(ap + 16);
        }

        // load ALL four B fragments first (one clause), then 4 WMMAs
        V16U Bf[4];
#pragma unroll
        for (int i = 0; i < 4; ++i) {
          const int arow = (w * 4 + i) * 16 + l15;
          const unsigned short* bp = WT + (size_t)arow * DE_ + kb + hi * 8;
          Bf[i].u[0] = *(const uint4*)bp;
          Bf[i].u[1] = *(const uint4*)(bp + 16);
        }
#pragma unroll
        for (int i = 0; i < 4; ++i) {
          acc[i] = __builtin_amdgcn_wmma_f32_16x16x32_bf16(
              false, A.v, false, Bf[i].v, (short)0, acc[i], false, false);
        }
      }
    }

    // Epilogue: score[p] = sum_a relu(att1 + pre) * Wfull[a]
    // C layout: lane 0-15 -> (M=r, N=lane); lane 16-31 -> (M=r+8, N=lane-16)
    float t[8];
#pragma unroll
    for (int r = 0; r < 8; ++r) {
      float s = 0.f;
#pragma unroll
      for (int i = 0; i < 4; ++i) {
        float v = acc[i][r] + av[i];
        v = v > 0.f ? v : 0.f;
        s += v * wv[i];
      }
      t[r] = s;
    }
    // reduce over N within each 16-lane half
#pragma unroll
    for (int m = 1; m < 16; m <<= 1) {
#pragma unroll
      for (int r = 0; r < 8; ++r) t[r] += __shfl_xor(t[r], m, 32);
    }
    if (l15 == 0) {
#pragma unroll
      for (int r = 0; r < 8; ++r) sPart[w][hi * 8 + r] = t[r];
    }
    __syncthreads();
    if (tid < 16) {
      float s = 0.f;
#pragma unroll
      for (int ww = 0; ww < 8; ++ww) s += sPart[ww][tid];
      const int p = p0 + tid;
      if (p < P_) sAtt[p] = s + bf_bias;
    }
  }

  // ---------------- softmax over P=196 ----------------
  __syncthreads();
  const float v = (tid < P_) ? sAtt[tid] : -3.0e38f;
  sRed[tid] = v;
  __syncthreads();
  for (int s = 128; s > 0; s >>= 1) {
    if (tid < s) sRed[tid] = fmaxf(sRed[tid], sRed[tid + s]);
    __syncthreads();
  }
  const float mx = sRed[0];
  __syncthreads();
  const float e = (tid < P_) ? __expf(v - mx) : 0.f;
  sRed[tid] = e;
  __syncthreads();
  for (int s = 128; s > 0; s >>= 1) {
    if (tid < s) sRed[tid] += sRed[tid + s];
    __syncthreads();
  }
  const float inv = 1.f / sRed[0];
  const float al = e * inv;
  if (tid < P_) {
    alpha[(size_t)b * P_ + tid] = al;
    sAtt[tid] = al;
  }
  __syncthreads();

  // ---------------- context[b,:] = sum_p alpha[p] * enc[b,p,:] --------------
  // slab is L2-hot from the GEMM pass; 8 f32 per thread, fully coalesced.
  float a0 = 0, a1 = 0, a2 = 0, a3 = 0, a4 = 0, a5 = 0, a6 = 0, a7 = 0;
  for (int p = 0; p < P_; ++p) {
    const float ap = sAtt[p];
    const float4* rp = (const float4*)(encB + (size_t)p * DE_) + tid * 2;
    const float4 x = rp[0];
    const float4 y = rp[1];
    a0 += ap * x.x; a1 += ap * x.y; a2 += ap * x.z; a3 += ap * x.w;
    a4 += ap * y.x; a5 += ap * y.y; a6 += ap * y.z; a7 += ap * y.w;
  }
  float* cp = ctx + (size_t)b * DE_ + tid * 8;
  *(float4*)cp       = make_float4(a0, a1, a2, a3);
  *(float4*)(cp + 4) = make_float4(a4, a5, a6, a7);
}

// ---------------------------------------------------------------------------
extern "C" void kernel_launch(void* const* d_in, const int* in_sizes, int n_in,
                              void* d_out, int out_size, void* d_ws,
                              size_t ws_size, hipStream_t stream) {
  const float* enc   = (const float*)d_in[0];  // [B,P,DE]
  const float* dh    = (const float*)d_in[1];  // [B,DD]
  const float* Wenc  = (const float*)d_in[2];  // [DE,DA]
  const float* benc  = (const float*)d_in[3];  // [DA]
  const float* Wdec  = (const float*)d_in[4];  // [DD,DA]
  const float* bdec  = (const float*)d_in[5];  // [DA]
  const float* Wfull = (const float*)d_in[6];  // [DA,1]
  const float* bfull = (const float*)d_in[7];  // [1]

  float* ctx   = (float*)d_out;                 // [B,DE]
  float* alpha = ctx + (size_t)B_ * DE_;        // [B,P]

  unsigned short* WT = (unsigned short*)d_ws;                       // 2 MB
  float* pre = (float*)((char*)d_ws + (size_t)DA_ * DE_ * 2);       // 512 KB

  wenc_transpose_bf16<<<dim3(DA_ / 32, DE_ / 32), 256, 0, stream>>>(Wenc, WT);
  att2_pre_kernel<<<(B_ * DA_) / 256, 256, 0, stream>>>(dh, Wdec, bdec, benc, pre);
  attn_main<<<B_, 256, 0, stream>>>(enc, WT, pre, Wfull, bfull, ctx, alpha);
}
